// EfficientAttention_1726576854490
// MI455X (gfx1250) — matrix operand order repaired and checked by reading, not compiled, not run on hardware
//
#include <hip/hip_runtime.h>
#include <hip/hip_bf16.h>
#include <stdint.h>
#include <stddef.h>

// ---------------------------------------------------------------------------
// EfficientAttention (linear attention) for MI455X / gfx1250, wave32 + WMMA.
//   k,q,v = W{k,q,v} @ x + b      (bf16 WMMA GEMM, 512x512 @ 512x(16*4096))
//   k -> softmax over spatial; q -> softmax over per-head channels
//   ctx = k @ v^T per (b,head); av = ctx^T @ q; out = Wr @ av + br
// All matmuls use v_wmma_f32_16x16x32_bf16 (f32 accumulate).
// Workspace: 4 bf16 tensors of 16*512*4096 elements = 268 MB.
// ---------------------------------------------------------------------------

typedef __attribute__((ext_vector_type(16))) __bf16        v16bf;
typedef __attribute__((ext_vector_type(2)))  __bf16        v2bf;
typedef __attribute__((ext_vector_type(8)))  float         v8f;
typedef __attribute__((ext_vector_type(8)))  unsigned int  v8u;

#define NHEADS 8
constexpr int Cc  = 512;    // channels
constexpr int HWc = 4096;   // spatial
constexpr int Bc  = 16;     // batch
constexpr int HK  = 64;     // per-head channels (hk == hv)

// ---- bf16 helpers: use native hardware converts (let clang pick v_cvt_*) ----
__device__ __forceinline__ unsigned pack_bf16(float a, float b) {
  v2bf r;
  r.x = (__bf16)a;
  r.y = (__bf16)b;
  return __builtin_bit_cast(unsigned, r);
}
__device__ __forceinline__ unsigned short f2bf(float f) {
  __bf16 h = (__bf16)f;
  return __builtin_bit_cast(unsigned short, h);
}
__device__ __forceinline__ float bf2f(unsigned short h) {
  return __builtin_bit_cast(float, (unsigned)h << 16);
}

// ---- WMMA fragment loaders (contiguous K order k = 16*(L/16) + s; A and B match) ----
// A-matrix 16x32 bf16: lane L: M = L%16; VGPR v holds K pair at
//   K = 2v + 16*(L/16).  Pairs contiguous along K.
__device__ __forceinline__ v16bf frag_ld_A(const unsigned short* p, int hl) {
  v8u b;
#pragma unroll
  for (int v = 0; v < 8; ++v) {
    int kb = 2 * v + 16 * hl;
    b[v] = *(const unsigned int*)(p + kb);
  }
  return __builtin_bit_cast(v16bf, b);
}
// B-matrix 32x16 bf16: lane L: N = L%16; VGPR v holds K pair at
//   K = 2v + 16*(L/16).  p points at (K=0, this lane's N) in a layout where
//   consecutive K are contiguous in memory (i.e. B stored N-major).
__device__ __forceinline__ v16bf frag_ld_B(const unsigned short* p, int hl) {
  v8u b;
#pragma unroll
  for (int v = 0; v < 8; ++v) {
    int kb = 2 * v + 16 * hl;
    b[v] = *(const unsigned int*)(p + kb);
  }
  return __builtin_bit_cast(v16bf, b);
}
// B fragment when K rows are strided by `ld` elements (K-major source).
__device__ __forceinline__ v16bf frag_ld_B_strided(const unsigned short* p,
                                                   int ld, int hl) {
  v8u b;
#pragma unroll
  for (int v = 0; v < 8; ++v) {
    int k0 = 2 * v + 16 * hl;
    unsigned lo = p[(size_t)k0 * ld];
    unsigned hi = p[(size_t)(k0 + 1) * ld];
    b[v] = lo | (hi << 16);
  }
  return __builtin_bit_cast(v16bf, b);
}

__device__ __forceinline__ v8f wmma_bf16(v16bf a, v16bf b, v8f c) {
  return __builtin_amdgcn_wmma_f32_16x16x32_bf16(
      false, a, false, b, (short)0, c, false, false);
}

// ---------------------------------------------------------------------------
// Kernel 1: fused K/Q/V projection GEMM.  P = W @ x + b, bf16 output.
// Tile: BM=64 (out ch) x BN=128 (spatial) x BK=32.  256 threads = 8 waves.
// Wave w owns the 16-wide column strip n = w*16, all 4 row subtiles.
// All LDS staging stores are packed 32-bit (bf16 pairs along K).
// ---------------------------------------------------------------------------
constexpr int BM = 64, BN = 128, BK = 32;
constexpr int LDW = BK + 2;  // even -> dword-aligned packed pairs
constexpr int LDX = BK + 2;

__global__ __launch_bounds__(256) void proj_kernel(
    const float* __restrict__ x,
    const float* __restrict__ Wk, const float* __restrict__ bk,
    const float* __restrict__ Wq, const float* __restrict__ bq,
    const float* __restrict__ Wv, const float* __restrict__ bv,
    unsigned short* __restrict__ Kb, unsigned short* __restrict__ Qb,
    unsigned short* __restrict__ Vb) {
  const int p = blockIdx.z % 3;
  const int b = blockIdx.z / 3;
  const float* W    = (p == 0) ? Wk : (p == 1) ? Wq : Wv;
  const float* bias = (p == 0) ? bk : (p == 1) ? bq : bv;
  unsigned short* out = (p == 0) ? Kb : (p == 1) ? Qb : Vb;

  __shared__ unsigned short ldsW[BM * LDW];  // [m][k]  (K-contiguous: A frags)
  __shared__ unsigned short ldsX[BN * LDX];  // [n][k]  (K-contiguous: B frags)

  const int tid = threadIdx.x;
  const int w = tid >> 5, lane = tid & 31, l16 = lane & 15, hl = lane >> 4;
  const int m0 = blockIdx.y * BM;
  const int n0 = blockIdx.x * BN;

  v8f acc[4] = {};

  for (int kk0 = 0; kk0 < Cc; kk0 += BK) {
    {  // stage 64x32 weight tile, f32 -> bf16 packed pairs along k
      int m = tid >> 2, kq = (tid & 3) * 8;
      const float* wp = &W[(size_t)(m0 + m) * Cc + kk0 + kq];
      if (kk0 + BK < Cc) __builtin_prefetch(wp + BK, 0, 1);
#pragma unroll
      for (int i = 0; i < 8; i += 2)
        *(unsigned*)&ldsW[m * LDW + kq + i] = pack_bf16(wp[i], wp[i + 1]);
    }
    {  // stage 32x128 x tile transposed ([n][k]); each thread reads two
       // adjacent channel rows (c, c+1) and writes packed dwords.
      int cp = (tid >> 4) * 2;        // 0,2,...,30
      int lq = (tid & 15) * 8;        // 0,8,...,120
      const float* x0 = &x[((size_t)b * Cc + kk0 + cp) * HWc + n0 + lq];
      const float* x1 = x0 + HWc;
      if (kk0 + BK < Cc) __builtin_prefetch(x0 + (size_t)BK * HWc, 0, 1);
#pragma unroll
      for (int i = 0; i < 8; ++i)
        *(unsigned*)&ldsX[(lq + i) * LDX + cp] = pack_bf16(x0[i], x1[i]);
    }
    __syncthreads();

    v16bf bf = frag_ld_B(&ldsX[(w * 16 + l16) * LDX], hl);
#pragma unroll
    for (int ms = 0; ms < 4; ++ms) {
      v16bf af = frag_ld_A(&ldsW[(ms * 16 + l16) * LDW], hl);
      acc[ms] = wmma_bf16(af, bf, acc[ms]);
    }
    __syncthreads();
  }

  // epilogue: bias add, bf16 store.  C layout: VGPR g -> M=g+8*hl, N=l16.
#pragma unroll
  for (int ms = 0; ms < 4; ++ms) {
#pragma unroll
    for (int g = 0; g < 8; ++g) {
      int row = m0 + ms * 16 + g + 8 * hl;
      float val = acc[ms][g] + bias[row];
      out[((size_t)b * Cc + row) * HWc + n0 + w * 16 + l16] = f2bf(val);
    }
  }
}

// ---------------------------------------------------------------------------
// Kernel 2: softmax over spatial axis (4096) for K, in place on bf16.
// One block per (b, channel) row; 256 threads x 16 values each, in registers.
// ---------------------------------------------------------------------------
__global__ __launch_bounds__(256) void softmax_k_kernel(
    unsigned short* __restrict__ Kb) {
  unsigned* rp = (unsigned*)(Kb + (size_t)blockIdx.x * HWc);
  const int tid = threadIdx.x, lane = tid & 31, wid = tid >> 5;
  __shared__ float redmax[8];
  __shared__ float redsum[8];

  unsigned u[8];
  float v[16];
#pragma unroll
  for (int i = 0; i < 8; ++i) u[i] = rp[tid * 8 + i];
#pragma unroll
  for (int i = 0; i < 8; ++i) {
    v[2 * i]     = bf2f((unsigned short)(u[i] & 0xffffu));
    v[2 * i + 1] = bf2f((unsigned short)(u[i] >> 16));
  }
  float mx = -3.4e38f;
#pragma unroll
  for (int i = 0; i < 16; ++i) mx = fmaxf(mx, v[i]);
#pragma unroll
  for (int off = 16; off >= 1; off >>= 1) mx = fmaxf(mx, __shfl_xor(mx, off, 32));
  if (lane == 0) redmax[wid] = mx;
  __syncthreads();
  mx = -3.4e38f;
#pragma unroll
  for (int j = 0; j < 8; ++j) mx = fmaxf(mx, redmax[j]);

  float s = 0.f;
#pragma unroll
  for (int i = 0; i < 16; ++i) { v[i] = __expf(v[i] - mx); s += v[i]; }
#pragma unroll
  for (int off = 16; off >= 1; off >>= 1) s += __shfl_xor(s, off, 32);
  if (lane == 0) redsum[wid] = s;
  __syncthreads();
  s = 0.f;
#pragma unroll
  for (int j = 0; j < 8; ++j) s += redsum[j];
  float inv = 1.f / s;

#pragma unroll
  for (int i = 0; i < 8; ++i)
    rp[tid * 8 + i] = pack_bf16(v[2 * i] * inv, v[2 * i + 1] * inv);
}

// ---------------------------------------------------------------------------
// Kernel 3: softmax over per-head channel axis (64) for Q, in place on bf16.
// Grid = B*HEADS*16 blocks; thread handles one spatial position, 64 channel
// values held in VGPRs.  Loads/stores coalesced along l.
// ---------------------------------------------------------------------------
__global__ __launch_bounds__(256) void softmax_q_kernel(
    unsigned short* __restrict__ Qb) {
  const int blk = blockIdx.x;
  const int b = blk / (NHEADS * 16);
  const int h = (blk / 16) % NHEADS;
  const int l = (blk % 16) * 256 + threadIdx.x;
  unsigned short* base = Qb + ((size_t)b * Cc + h * HK) * HWc + l;

  float vals[HK];
#pragma unroll
  for (int c = 0; c < HK; ++c) vals[c] = bf2f(base[(size_t)c * HWc]);
  float mx = -3.4e38f;
#pragma unroll
  for (int c = 0; c < HK; ++c) mx = fmaxf(mx, vals[c]);
  float s = 0.f;
#pragma unroll
  for (int c = 0; c < HK; ++c) { vals[c] = __expf(vals[c] - mx); s += vals[c]; }
  float inv = 1.f / s;
#pragma unroll
  for (int c = 0; c < HK; ++c) base[(size_t)c * HWc] = f2bf(vals[c] * inv);
}

// ---------------------------------------------------------------------------
// Kernel 4: per-(b,head) fused  ctx = K' V^T  then  av = ctx^T Q'.
// Phase 1: 64x64 GEMM, K-dim 4096, fragments loaded directly from global
//   (both A pairs and B pairs are contiguous along l).  8 waves: wave w owns
//   row subtile w&3 and 32-wide column strip (w>>2)*32.
// Phase 2: ctx^T staged in LDS (bf16, packed dword stores); 64x4096 GEMM.
// ---------------------------------------------------------------------------
__global__ __launch_bounds__(256) void ctx_av_kernel(
    const unsigned short* __restrict__ Kb,
    const unsigned short* __restrict__ Qb,
    const unsigned short* __restrict__ Vb,
    unsigned short* __restrict__ AGb) {
  const int b = blockIdx.x >> 3;
  const int h = blockIdx.x & 7;
  const unsigned short* Kh = Kb + ((size_t)b * Cc + h * HK) * HWc;
  const unsigned short* Qh = Qb + ((size_t)b * Cc + h * HK) * HWc;
  const unsigned short* Vh = Vb + ((size_t)b * Cc + h * HK) * HWc;
  unsigned short* Ah = AGb + ((size_t)b * Cc + h * HK) * HWc;

  const int tid = threadIdx.x;
  const int w = tid >> 5, lane = tid & 31, l16 = lane & 15, hl = lane >> 4;

  constexpr int CTS = 66;                    // even stride, padded
  __shared__ unsigned short ctxT[HK * CTS];  // [hv][hk] bf16

  // ---- phase 1: ctx[m=hk][n=hv] = sum_l K'[m,l] * V[n,l] ----
  const int ms = w & 3;
  const int nb = (w >> 2) * 32;
  v8f a0 = {}, a1 = {};
  for (int l0 = 0; l0 < HWc; l0 += 32) {
    v16bf af = frag_ld_A(Kh + (size_t)(ms * 16 + l16) * HWc + l0, hl);
    v16bf b0 = frag_ld_B(Vh + (size_t)(nb + l16) * HWc + l0, hl);
    v16bf b1 = frag_ld_B(Vh + (size_t)(nb + 16 + l16) * HWc + l0, hl);
    a0 = wmma_bf16(af, b0, a0);
    a1 = wmma_bf16(af, b1, a1);
  }
  // store ctx^T to LDS; consecutive accumulator elements are consecutive M
  // (same row in ctxT) -> packed dword stores.  M base is even.
#pragma unroll
  for (int g = 0; g < 8; g += 2) {
    int m = ms * 16 + g + 8 * hl;              // hk index (even)
    *(unsigned*)&ctxT[(nb + l16) * CTS + m]      = pack_bf16(a0[g], a0[g + 1]);
    *(unsigned*)&ctxT[(nb + 16 + l16) * CTS + m] = pack_bf16(a1[g], a1[g + 1]);
  }
  __syncthreads();

  // ---- phase 2: av[m=hv][n=l] = sum_k ctxT[m,k] * Q'[k,n] ----
  v16bf afr[4][2];
#pragma unroll
  for (int mv = 0; mv < 4; ++mv)
#pragma unroll
    for (int kk = 0; kk < 2; ++kk)
      afr[mv][kk] = frag_ld_A(&ctxT[(mv * 16 + l16) * CTS + kk * 32], hl);

  for (int i = 0; i < 32; ++i) {
    int n0 = (w + 8 * i) * 16;
    v16bf bf0 = frag_ld_B_strided(Qh + n0 + l16, HWc, hl);                    // k 0..31
    v16bf bf1 = frag_ld_B_strided(Qh + (size_t)32 * HWc + n0 + l16, HWc, hl); // k 32..63
#pragma unroll
    for (int mv = 0; mv < 4; ++mv) {
      v8f acc = {};
      acc = wmma_bf16(afr[mv][0], bf0, acc);
      acc = wmma_bf16(afr[mv][1], bf1, acc);
#pragma unroll
      for (int g = 0; g < 8; ++g) {
        int m = mv * 16 + g + 8 * hl;          // hv index
        Ah[(size_t)m * HWc + n0 + l16] = f2bf(acc[g]);
      }
    }
  }
}

// ---------------------------------------------------------------------------
// Kernel 5: output projection.  out = Wr @ agg + br  (bf16 in, f32 out).
// Same tiling as proj_kernel; X staging packs two bf16 rows per dword.
// ---------------------------------------------------------------------------
__global__ __launch_bounds__(256) void rproj_kernel(
    const unsigned short* __restrict__ agg,
    const float* __restrict__ Wr, const float* __restrict__ br,
    float* __restrict__ out) {
  const int b = blockIdx.z;
  __shared__ unsigned short ldsW[BM * LDW];
  __shared__ unsigned short ldsX[BN * LDX];

  const int tid = threadIdx.x;
  const int w = tid >> 5, lane = tid & 31, l16 = lane & 15, hl = lane >> 4;
  const int m0 = blockIdx.y * BM;
  const int n0 = blockIdx.x * BN;

  v8f acc[4] = {};

  for (int kk0 = 0; kk0 < Cc; kk0 += BK) {
    {
      int m = tid >> 2, kq = (tid & 3) * 8;
      const float* wp = &Wr[(size_t)(m0 + m) * Cc + kk0 + kq];
      if (kk0 + BK < Cc) __builtin_prefetch(wp + BK, 0, 1);
#pragma unroll
      for (int i = 0; i < 8; i += 2)
        *(unsigned*)&ldsW[m * LDW + kq + i] = pack_bf16(wp[i], wp[i + 1]);
    }
    {
      int cp = (tid >> 4) * 2;
      int lq = (tid & 15) * 8;
      const unsigned short* a0 =
          &agg[((size_t)b * Cc + kk0 + cp) * HWc + n0 + lq];
      const unsigned short* a1 = a0 + HWc;
      if (kk0 + BK < Cc) __builtin_prefetch(a0 + (size_t)BK * HWc, 0, 1);
#pragma unroll
      for (int i = 0; i < 8; ++i)
        *(unsigned*)&ldsX[(lq + i) * LDX + cp] =
            (unsigned)a0[i] | ((unsigned)a1[i] << 16);
    }
    __syncthreads();

    v16bf bf = frag_ld_B(&ldsX[(w * 16 + l16) * LDX], hl);
#pragma unroll
    for (int ms = 0; ms < 4; ++ms) {
      v16bf af = frag_ld_A(&ldsW[(ms * 16 + l16) * LDW], hl);
      acc[ms] = wmma_bf16(af, bf, acc[ms]);
    }
    __syncthreads();
  }

#pragma unroll
  for (int ms = 0; ms < 4; ++ms) {
#pragma unroll
    for (int g = 0; g < 8; ++g) {
      int row = m0 + ms * 16 + g + 8 * hl;
      float val = acc[ms][g] + br[row];
      out[((size_t)b * Cc + row) * HWc + n0 + w * 16 + l16] = val;
    }
  }
}

// ---------------------------------------------------------------------------
extern "C" void kernel_launch(void* const* d_in, const int* in_sizes, int n_in,
                              void* d_out, int out_size, void* d_ws,
                              size_t ws_size, hipStream_t stream) {
  (void)in_sizes; (void)n_in; (void)out_size; (void)ws_size;
  const float* x  = (const float*)d_in[0];
  const float* Wk = (const float*)d_in[1];
  const float* bk = (const float*)d_in[2];
  const float* Wq = (const float*)d_in[3];
  const float* bq = (const float*)d_in[4];
  const float* Wv = (const float*)d_in[5];
  const float* bv = (const float*)d_in[6];
  const float* Wr = (const float*)d_in[7];
  const float* br = (const float*)d_in[8];
  float* out = (float*)d_out;

  // workspace: 4 bf16 tensors of B*C*HW elements (64 MB each, 256 MB total)
  const size_t S = (size_t)Bc * Cc * HWc;
  unsigned short* Kb  = (unsigned short*)d_ws;
  unsigned short* Qb  = Kb + S;
  unsigned short* Vb  = Qb + S;
  unsigned short* AGb = Vb + S;

  proj_kernel<<<dim3(HWc / BN, Cc / BM, 3 * Bc), dim3(256), 0, stream>>>(
      x, Wk, bk, Wq, bq, Wv, bv, Kb, Qb, Vb);
  softmax_k_kernel<<<dim3(Bc * Cc), dim3(256), 0, stream>>>(Kb);
  softmax_q_kernel<<<dim3(Bc * NHEADS * 16), dim3(256), 0, stream>>>(Qb);
  ctx_av_kernel<<<dim3(Bc * NHEADS), dim3(256), 0, stream>>>(Kb, Qb, Vb, AGb);
  rproj_kernel<<<dim3(HWc / BN, Cc / BM, Bc), dim3(256), 0, stream>>>(
      AGb, Wr, br, out);
}
